// PointNetPlusEncoderMSG_63874753626494
// MI455X (gfx1250) — matrix operand order repaired and checked by reading, not compiled, not run on hardware
//
#include <hip/hip_runtime.h>

typedef __attribute__((ext_vector_type(16))) _Float16 v16h;
typedef __attribute__((ext_vector_type(8)))  float    v8f;

// ============================================================================
// Farthest-point sampling: one block per batch. dist[] in LDS; argmax via
// wave32 shuffle reduction + tiny cross-wave LDS stage (2 barriers/iter).
// Lowest-index tie-break matches jnp.argmax first-hit.
// ============================================================================
__global__ void fps_kernel(const float* __restrict__ xyz, int N, int npoint,
                           float* __restrict__ new_xyz)
{
    extern __shared__ char dynsm[];
    float* dist = (float*)dynsm;                  // N floats
    float* rv   = dist + N;                       // nwaves floats
    int*   ri   = (int*)(rv + (blockDim.x >> 5)); // nwaves ints
    int*   sel  = ri + (blockDim.x >> 5);         // npoint ints

    const int b   = blockIdx.x;
    const int tid = threadIdx.x;
    const int lane = tid & 31, wave = tid >> 5, nwaves = blockDim.x >> 5;
    const float* p = xyz + (size_t)b * N * 3;

    for (int i = tid; i < N; i += blockDim.x) dist[i] = 1e10f;
    if (tid == 0) sel[0] = 0;
    __syncthreads();

    int last = 0;
    for (int it = 1; it < npoint; ++it) {
        const float lx = p[last*3+0], ly = p[last*3+1], lz = p[last*3+2];
        float bestv = -1.0f; int besti = 0x7fffffff;
        for (int i = tid; i < N; i += blockDim.x) {
            const float dx = p[i*3+0]-lx, dy = p[i*3+1]-ly, dz = p[i*3+2]-lz;
            float d = dx*dx + dy*dy + dz*dz;
            const float old = dist[i];
            d = d < old ? d : old;
            dist[i] = d;
            if (d > bestv || (d == bestv && i < besti)) { bestv = d; besti = i; }
        }
        for (int o = 16; o > 0; o >>= 1) {
            const float v2 = __shfl_down(bestv, o);
            const int   i2 = __shfl_down(besti, o);
            if (v2 > bestv || (v2 == bestv && i2 < besti)) { bestv = v2; besti = i2; }
        }
        if (lane == 0) { rv[wave] = bestv; ri[wave] = besti; }
        __syncthreads();
        if (tid == 0) {
            float bv = rv[0]; int bi = ri[0];
            for (int w = 1; w < nwaves; ++w)
                if (rv[w] > bv || (rv[w] == bv && ri[w] < bi)) { bv = rv[w]; bi = ri[w]; }
            sel[it] = bi;
        }
        __syncthreads();
        last = sel[it];
    }
    for (int i = tid; i < npoint * 3; i += blockDim.x) {
        const int s = i / 3, c = i - s * 3;
        new_xyz[((size_t)b * npoint + s) * 3 + c] = p[sel[s]*3 + c];
    }
}

// ============================================================================
// Ball query: one wave32 per centroid; ballot+popc compaction reproduces
// sort(where(mask, idx, n))[:nsample] + pad-with-first (0 if empty).
// ============================================================================
__global__ void ballq_kernel(const float* __restrict__ xyz, int N,
                             const float* __restrict__ ctr, int S, int BS,
                             float r2, int ns, int* __restrict__ oidx)
{
    const int gw = (int)((blockIdx.x * blockDim.x + threadIdx.x) >> 5);
    if (gw >= BS) return;                          // uniform per wave
    const int lane = threadIdx.x & 31;
    const int b = gw / S, s = gw - b * S;
    const float* p = xyz + (size_t)b * N * 3;
    const float* c = ctr + ((size_t)b * S + s) * 3;
    const float cx = c[0], cy = c[1], cz = c[2];
    int* o = oidx + ((size_t)b * S + s) * ns;

    int count = 0, firstIdx = 0;
    for (int base = 0; base < N && count < ns; base += 32) {
        const int i = base + lane;
        bool ok = false;
        if (i < N) {
            const float dx = p[i*3+0]-cx, dy = p[i*3+1]-cy, dz = p[i*3+2]-cz;
            ok = (dx*dx + dy*dy + dz*dz) < r2;
        }
        const unsigned mask = (unsigned)__ballot(ok);
        if (count == 0 && mask) firstIdx = base + __builtin_ctz(mask);
        if (ok) {
            const int pos = count + __popc(mask & ((1u << lane) - 1u));
            if (pos < ns) o[pos] = i;
        }
        count += __popc(mask);
    }
    if (count < ns) {
        const int pad = (count > 0) ? firstIdx : 0;
        for (int t = count + lane; t < ns; t += 32) o[t] = pad;
    }
}

// ============================================================================
// Weight prep: W (Kin x Cout f32) -> transposed padded f16 wt[n*Kpad + k].
// ============================================================================
__global__ void prep_w_kernel(const float* __restrict__ W, int Kin, int Cout,
                              int Kpad, _Float16* __restrict__ wt)
{
    const int i = blockIdx.x * blockDim.x + threadIdx.x;
    if (i >= Cout * Kpad) return;
    const int n = i / Kpad, k = i - n * Kpad;
    wt[i] = (_Float16)((k < Kin) ? W[(size_t)k * Cout + n] : 0.0f);
}

// ============================================================================
// Fragment loaders (contiguous k = 16h + s order, same in A and B).
// ============================================================================
__device__ __forceinline__ void loadA(v16h& a, const _Float16* __restrict__ arow,
                                      int kb, int h)
{
#pragma unroll
    for (int v = 0; v < 8; ++v) {
        // lane m=L%16, h=L/16; pair v holds K = h*16 + v*2
        const int kk = kb + (h << 4) + (v << 1);
        a[2*v]   = arow[kk];
        a[2*v+1] = arow[kk + 1];
    }
}
__device__ __forceinline__ void loadB(v16h& bf, const _Float16* __restrict__ wrow, int kb)
{
#pragma unroll
    for (int j = 0; j < 16; ++j) bf[j] = wrow[kb + j];  // N=L%16, K=(L/16)*16+j
}

// ============================================================================
// Register-blocked (1 M-tile x 2 N-tiles per wave), software-pipelined K-loop
// with peeled epilogue: in-loop prefetch is unconditional (no zero-init, no
// per-iteration branch). A from LDS, B from global transposed weights.
// ============================================================================
__device__ __forceinline__ void gemm_relu(const _Float16* __restrict__ inb, int Kpad,
                                          const _Float16* __restrict__ wt,
                                          const float* __restrict__ bias,
                                          int Cout, _Float16* __restrict__ outb,
                                          int ns, int lane, int wave, int nwaves)
{
    const int nM = ns >> 4, nP = Cout >> 5;   // pairs of 16-col tiles (Cout % 32 == 0)
    const int m = lane & 15;
    const int h = lane >> 4;
    for (int t = wave; t < nM * nP; t += nwaves) {
        const int mt = t / nP, np = t - mt * nP;
        const int n0 = np << 5;
        v8f acc0 = {}, acc1 = {};
        const _Float16* arow  = inb + (size_t)((mt << 4) + m) * Kpad;
        const _Float16* wrow0 = wt  + (size_t)(n0 + m) * Kpad + (h << 4);
        const _Float16* wrow1 = wrow0 + (size_t)16 * Kpad;

        v16h aC, b0C, b1C;
        loadA(aC, arow, 0, h); loadB(b0C, wrow0, 0); loadB(b1C, wrow1, 0);
        for (int kb = 0; kb + 32 < Kpad; kb += 32) {
            v16h aN, b0N, b1N;                  // unconditionally filled
            loadA(aN, arow, kb + 32, h);
            loadB(b0N, wrow0, kb + 32);
            loadB(b1N, wrow1, kb + 32);
            acc0 = __builtin_amdgcn_wmma_f32_16x16x32_f16(
                false, aC, false, b0C, (short)0, acc0, false, false);
            acc1 = __builtin_amdgcn_wmma_f32_16x16x32_f16(
                false, aC, false, b1C, (short)0, acc1, false, false);
            aC = aN; b0C = b0N; b1C = b1N;
        }
        acc0 = __builtin_amdgcn_wmma_f32_16x16x32_f16(
            false, aC, false, b0C, (short)0, acc0, false, false);
        acc1 = __builtin_amdgcn_wmma_f32_16x16x32_f16(
            false, aC, false, b1C, (short)0, acc1, false, false);

        const float bv0 = bias[n0 + m];
        const float bv1 = bias[n0 + 16 + m];
#pragma unroll
        for (int r = 0; r < 8; ++r) {           // C: col=L%16, row=(L/16)*8+r
            const int mm = (mt << 4) + (h << 3) + r;
            const float v0 = acc0[r] + bv0;
            const float v1 = acc1[r] + bv1;
            outb[(size_t)mm * Cout + n0 + m]      = (_Float16)(v0 > 0.0f ? v0 : 0.0f);
            outb[(size_t)mm * Cout + n0 + 16 + m] = (_Float16)(v1 > 0.0f ? v1 : 0.0f);
        }
    }
}

// ============================================================================
// Fused per-group pipeline: gather -> 3x(GEMM+bias+ReLU) LDS ping-pong ->
// max-pool -> pooled row to global (f16 intermediate stages, f32 final).
// mode 0: feat = g_xyz (SA1). mode 1: feat = [pts, g_xyz] (SA2).
// mode 2: feat = [xyz, pts], idx identity, no centering (SA3).
// ============================================================================
__global__ void fused_branch_kernel(
    const float* __restrict__ xyz, int N,
    const _Float16* __restrict__ pts, int Cpts,
    const float* __restrict__ ctr,
    const int*   __restrict__ idx,
    void* __restrict__ outv, int out_f32, int out_stride, int out_choff,
    int S, int ns, int mode, int K0pad,
    const _Float16* __restrict__ w0, const float* __restrict__ b0, int C1,
    const _Float16* __restrict__ w1, const float* __restrict__ b1, int C2,
    const _Float16* __restrict__ w2, const float* __restrict__ b2, int C3,
    int buf1_off_halfs)
{
    extern __shared__ char dynsm[];
    _Float16* buf0 = (_Float16*)dynsm;
    _Float16* buf1 = buf0 + buf1_off_halfs;

    const int bi = blockIdx.x;
    const int b = bi / S, s = bi - b * S;
    const int tid = threadIdx.x;
    const int lane = tid & 31, wave = tid >> 5, nwaves = blockDim.x >> 5;

    const int* gidx = idx ? (idx + ((size_t)b * S + s) * ns) : nullptr;
    float cx = 0.f, cy = 0.f, cz = 0.f;
    if (ctr) { const float* c = ctr + ((size_t)b * S + s) * 3; cx = c[0]; cy = c[1]; cz = c[2]; }

    // ---- gather group features into buf0 (ns x K0pad, f16, zero padded) ----
    const int tot = ns * K0pad;
    for (int i = tid; i < tot; i += blockDim.x) {
        const int j = i / K0pad, c = i - j * K0pad;
        const int src = gidx ? gidx[j] : j;
        _Float16 hv = (_Float16)0.0f;
        if (mode == 0) {
            if (c < 3)
                hv = (_Float16)(xyz[((size_t)b*N + src)*3 + c] - (c==0?cx:(c==1?cy:cz)));
        } else if (mode == 1) {
            if (c < Cpts) hv = pts[((size_t)b*N + src)*Cpts + c];
            else if (c < Cpts + 3) {
                const int cc = c - Cpts;
                hv = (_Float16)(xyz[((size_t)b*N + src)*3 + cc] - (cc==0?cx:(cc==1?cy:cz)));
            }
        } else {
            if (c < 3) hv = (_Float16)xyz[((size_t)b*N + src)*3 + c];
            else if (c < 3 + Cpts) hv = pts[((size_t)b*N + src)*Cpts + (c - 3)];
        }
        buf0[(size_t)j * K0pad + c] = hv;
    }
    __syncthreads();

    gemm_relu(buf0, K0pad, w0, b0, C1, buf1, ns, lane, wave, nwaves);
    __syncthreads();
    gemm_relu(buf1, C1,    w1, b1, C2, buf0, ns, lane, wave, nwaves);
    __syncthreads();
    gemm_relu(buf0, C2,    w2, b2, C3, buf1, ns, lane, wave, nwaves);
    __syncthreads();

    // ---- max-pool over the ns samples ----
    for (int c = tid; c < C3; c += blockDim.x) {
        float mx = -1e30f;
        for (int j = 0; j < ns; ++j) {
            const float v = (float)buf1[(size_t)j * C3 + c];
            mx = v > mx ? v : mx;
        }
        const size_t ofs = ((size_t)b * S + s) * out_stride + out_choff + c;
        if (out_f32) ((float*)outv)[ofs] = mx;
        else         ((_Float16*)outv)[ofs] = (_Float16)mx;
    }
}

// ============================================================================
// Host orchestration
// ============================================================================
extern "C" void kernel_launch(void* const* d_in, const int* in_sizes, int n_in,
                              void* d_out, int out_size, void* d_ws, size_t ws_size,
                              hipStream_t stream)
{
    (void)in_sizes; (void)n_in; (void)out_size; (void)ws_size;
    const float* x = (const float*)d_in[0];

    // pytree order: x, sa1 b0..b2 (W,b)x3, sa2 b0..b2, sa3 (= "branch" 6)
    const float* W[7][3]; const float* Bv[7][3];
    { int p = 1;
      for (int br = 0; br < 7; ++br)
        for (int l = 0; l < 3; ++l) { W[br][l] = (const float*)d_in[p++]; Bv[br][l] = (const float*)d_in[p++]; } }

    static const int   K0S[7]    = {3,3,3,323,323,323,643};
    static const int   WID[7][3] = {{32,32,64},{64,64,128},{64,96,128},
                                    {64,64,128},{128,128,256},{128,128,256},
                                    {256,512,512}};
    static const int   NSS[6]    = {16,32,128,32,64,128};
    static const float RAD[6]    = {0.1f,0.2f,0.4f,0.2f,0.4f,0.8f};
    static const int   CHOFF[6]  = {0,64,192,0,128,384};

    char* ws = (char*)d_ws;
    size_t off = 0;
    auto alloc = [&](size_t bytes)->size_t {
        size_t o = off; off = (off + bytes + 255) & ~(size_t)255; return o; };

    size_t wtoff[7][3]; int kpad[7][3];
    for (int br = 0; br < 7; ++br)
        for (int l = 0; l < 3; ++l) {
            const int Kin = (l == 0) ? K0S[br] : WID[br][l-1];
            const int Kp  = (Kin + 31) & ~31;
            kpad[br][l]  = Kp;
            wtoff[br][l] = alloc((size_t)WID[br][l] * Kp * 2);
        }
    const size_t o_nx1 = alloc((size_t)8*512*3*4);
    const size_t o_l1  = alloc((size_t)8*512*320*2);   // f16
    const size_t o_nx2 = alloc((size_t)8*128*3*4);
    const size_t o_l2  = alloc((size_t)8*128*640*2);   // f16
    size_t o_bidx[6];
    for (int i = 0; i < 6; ++i) {
        const int S = (i < 3) ? 512 : 128;
        o_bidx[i] = alloc((size_t)8 * S * NSS[i] * 4);
    }

    // 1) convert+transpose+pad all weights to f16 pool
    for (int br = 0; br < 7; ++br)
        for (int l = 0; l < 3; ++l) {
            const int Kin = (l == 0) ? K0S[br] : WID[br][l-1];
            const int tot = WID[br][l] * kpad[br][l];
            prep_w_kernel<<<(tot + 255)/256, 256, 0, stream>>>(
                W[br][l], Kin, WID[br][l], kpad[br][l], (_Float16*)(ws + wtoff[br][l]));
        }

    // 2) FPS level 1: 8192 -> 512 centroids
    fps_kernel<<<8, 256, 8192*4 + 8*8 + 512*4, stream>>>(
        x, 8192, 512, (float*)(ws + o_nx1));

    // 3) SA1: ball query + fused group-MLP-maxpool per branch
    for (int br = 0; br < 3; ++br) {
        const int S = 512, ns = NSS[br], BS = 8 * S;
        ballq_kernel<<<BS/4, 128, 0, stream>>>(
            x, 8192, (const float*)(ws + o_nx1), S, BS,
            RAD[br]*RAD[br], ns, (int*)(ws + o_bidx[br]));
        const int C1 = WID[br][0], C2 = WID[br][1], C3 = WID[br][2], K0p = kpad[br][0];
        const int b1off = ((ns * (K0p > C2 ? K0p : C2)) + 15) & ~15;
        const size_t lds = (size_t)(b1off + ns * (C1 > C3 ? C1 : C3)) * 2;
        fused_branch_kernel<<<BS, 256, lds, stream>>>(
            x, 8192, nullptr, 0, (const float*)(ws + o_nx1), (const int*)(ws + o_bidx[br]),
            (void*)(ws + o_l1), 0, 320, CHOFF[br], S, ns, /*mode*/0, K0p,
            (const _Float16*)(ws + wtoff[br][0]), Bv[br][0], C1,
            (const _Float16*)(ws + wtoff[br][1]), Bv[br][1], C2,
            (const _Float16*)(ws + wtoff[br][2]), Bv[br][2], C3,
            b1off);
    }

    // 4) FPS level 2: 512 -> 128 centroids
    fps_kernel<<<8, 256, 512*4 + 8*8 + 128*4, stream>>>(
        (const float*)(ws + o_nx1), 512, 128, (float*)(ws + o_nx2));

    // 5) SA2 branches (points = l1_pts f16, 320 ch, then g_xyz)
    for (int br = 3; br < 6; ++br) {
        const int S = 128, ns = NSS[br], BS = 8 * S;
        ballq_kernel<<<BS/4, 128, 0, stream>>>(
            (const float*)(ws + o_nx1), 512, (const float*)(ws + o_nx2), S, BS,
            RAD[br]*RAD[br], ns, (int*)(ws + o_bidx[br]));
        const int C1 = WID[br][0], C2 = WID[br][1], C3 = WID[br][2], K0p = kpad[br][0]; // 352
        const int b1off = ((ns * (K0p > C2 ? K0p : C2)) + 15) & ~15;
        const size_t lds = (size_t)(b1off + ns * (C1 > C3 ? C1 : C3)) * 2;
        fused_branch_kernel<<<BS, 256, lds, stream>>>(
            (const float*)(ws + o_nx1), 512, (const _Float16*)(ws + o_l1), 320,
            (const float*)(ws + o_nx2), (const int*)(ws + o_bidx[br]),
            (void*)(ws + o_l2), 0, 640, CHOFF[br], S, ns, /*mode*/1, K0p,
            (const _Float16*)(ws + wtoff[br][0]), Bv[br][0], C1,
            (const _Float16*)(ws + wtoff[br][1]), Bv[br][1], C2,
            (const _Float16*)(ws + wtoff[br][2]), Bv[br][2], C3,
            b1off);
    }

    // 6) SA3: global MLP over 128 points ([l2_xyz, l2_pts] -> 643ch -> 512) + max
    {
        const int ns = 128, K0p = kpad[6][0];             // 672
        const int C1 = 256, C2 = 512, C3 = 512;
        const int b1off = ((ns * (K0p > C2 ? K0p : C2)) + 15) & ~15;  // 128*672
        const size_t lds = (size_t)(b1off + ns * (C1 > C3 ? C1 : C3)) * 2; // ~296 KB
        fused_branch_kernel<<<8, 256, lds, stream>>>(
            (const float*)(ws + o_nx2), 128, (const _Float16*)(ws + o_l2), 640,
            nullptr, nullptr,
            d_out, 1, 512, 0, /*S*/1, ns, /*mode*/2, K0p,
            (const _Float16*)(ws + wtoff[6][0]), Bv[6][0], C1,
            (const _Float16*)(ws + wtoff[6][1]), Bv[6][1], C2,
            (const _Float16*)(ws + wtoff[6][2]), Bv[6][2], C3,
            b1off);
    }
}
